// DiffusionOrderingNetwork_44117904065182
// MI455X (gfx1250) — matrix operand order repaired and checked by reading, not compiled, not run on hardware
//
#include <hip/hip_runtime.h>

// ---------------- problem constants ----------------
#define NN       50000
#define EE       800000
#define DD       256
#define RR       6
#define NLAYERS  3
#define KORD     50                  // node_order length
#define KTOT     (RR*DD + DD)        // 1792 combined K (relations ++ root)
#define NKC      (KTOT/32)           // 56 K-chunks of 32
#define NKC_REL  (RR*DD/32)          // 48 chunks from relation means
#define NKC_HEAD (DD/32)             // 8 K-chunks for head GEMM
#define PER_LAYER_FRAG (NKC*16*32*16)        // 458752 bf16 elems per layer
#define HEAD_FRAG      (NKC_HEAD*16*32*16)   // 65536 bf16 elems
#define TOT_FRAG       (NLAYERS*PER_LAYER_FRAG + HEAD_FRAG)
#define LDS_STRIDE 40    // bf16 elems per A-tile row (80B: 16B aligned, conflict-free)

typedef __attribute__((ext_vector_type(16))) __bf16 v16bf;
typedef __attribute__((ext_vector_type(8)))  float  v8f;

union FragBF {
  v16bf v;
  uint4 q[2];
};

__device__ __forceinline__ unsigned f2bf1(float x) {
  union { float f; unsigned u; } c; c.f = x;
  unsigned u = c.u;
  return (u + 0x7FFFu + ((u >> 16) & 1u)) >> 16;   // round-to-nearest-even
}
__device__ __forceinline__ unsigned f2bf_pack(float lo, float hi) {
  return f2bf1(lo) | (f2bf1(hi) << 16);
}

// ---------------- utility kernels ----------------
__global__ void k_zero4(float4* __restrict__ p, int n4) {
  int i = blockIdx.x * blockDim.x + threadIdx.x;
  if (i < n4) p[i] = make_float4(0.f, 0.f, 0.f, 0.f);
}

// h[n][d] = emb[x[n]][d]
__global__ void k_embed(const int* __restrict__ x, const float4* __restrict__ emb4,
                        float4* __restrict__ h4) {
  int i = blockIdx.x * blockDim.x + threadIdx.x;      // over N*64
  if (i >= NN * (DD / 4)) return;
  int n = i >> 6, c = i & 63;
  h4[i] = emb4[x[n] * (DD / 4) + c];
}

// h[node_order[i]] += pos[i]   (atomic: node_order may repeat)
__global__ void k_posadd(const int* __restrict__ ord, const float* __restrict__ pos,
                         float* __restrict__ h) {
  int i = blockIdx.x * blockDim.x + threadIdx.x;      // over KORD*DD
  if (i >= KORD * DD) return;
  int r = i >> 8, d = i & 255;
  atomicAdd(&h[(size_t)ord[r] * DD + d], pos[i]);
}

// pack W_rel/W_root/w1 into bf16 B-fragment-native layout
__global__ void k_prep(const float* __restrict__ W_rel, const float* __restrict__ W_root,
                       const float* __restrict__ w1, unsigned short* __restrict__ wf) {
  int idx = blockIdx.x * blockDim.x + threadIdx.x;
  if (idx >= TOT_FRAG) return;
  int i2, l; bool head;
  if (idx < NLAYERS * PER_LAYER_FRAG) { l = idx / PER_LAYER_FRAG; i2 = idx % PER_LAYER_FRAG; head = false; }
  else                                { l = 0; i2 = idx - NLAYERS * PER_LAYER_FRAG; head = true; }
  int j    = i2 & 15;
  int lane = (i2 >> 4) & 31;
  int ft   = (i2 >> 9) & 15;
  int kc   = i2 >> 13;
  int f = ft * 16 + (lane & 15);
  int k = kc * 32 + ((lane >> 4) << 4) + j;   // B lane group: K 0..15 / 16..31
  float v;
  if (head)            v = w1[k * DD + f];
  else if (k < RR*DD)  { int r = k >> 8, d = k & 255;
                         v = W_rel[(((size_t)l * RR + r) * DD + d) * DD + f]; }
  else                 v = W_root[((size_t)l * DD + (k - RR*DD)) * DD + f];
  wf[idx] = (unsigned short)f2bf1(v);
}

// edge counts per (dst, relation) segment
__global__ void k_count(const int* __restrict__ dst, const int* __restrict__ et,
                        float* __restrict__ cnt) {
  int e = blockIdx.x * blockDim.x + threadIdx.x;
  if (e >= EE) return;
  atomicAdd(&cnt[dst[e] * RR + et[e]], 1.0f);
}

// cnt -> 1/max(cnt,1)  (hoists all division out of the GEMM staging loop)
__global__ void k_rcnt(float* __restrict__ cnt) {
  int i = blockIdx.x * blockDim.x + threadIdx.x;
  if (i >= NN * RR) return;
  cnt[i] = 1.0f / fmaxf(cnt[i], 1.0f);
}

// one wave per edge: sums[dst*R+et] += h[src]   (fp32 atomics, coalesced row move)
__global__ void k_scatter(const int* __restrict__ src, const int* __restrict__ dst,
                          const int* __restrict__ et, const float* __restrict__ h,
                          float* __restrict__ sums) {
  int t = blockIdx.x * blockDim.x + threadIdx.x;
  int lane = t & 31;
  int e = __builtin_amdgcn_readfirstlane(t >> 5);   // wave-uniform -> SGPR, s_load paths
  const float4* in4 = (const float4*)(h + (size_t)src[e] * DD);
  float* out = sums + ((size_t)dst[e] * RR + et[e]) * DD;
  float4 a = in4[lane];
  float4 b = in4[lane + 32];
  int d0 = lane * 4;
  atomicAdd(&out[d0 + 0], a.x); atomicAdd(&out[d0 + 1], a.y);
  atomicAdd(&out[d0 + 2], a.z); atomicAdd(&out[d0 + 3], a.w);
  atomicAdd(&out[128 + d0 + 0], b.x); atomicAdd(&out[128 + d0 + 1], b.y);
  atomicAdd(&out[128 + d0 + 2], b.z); atomicAdd(&out[128 + d0 + 3], b.w);
}

// ---------------- fused RGCN layer GEMM (bf16 WMMA, double-buffered LDS) ----
// block = 256 thr (8 waves), computes rows [n0, n0+16) x all 256 out features.
// A[n, k] = sums*rcnt (k<1536) or h[n, k-1536]; B = prepacked fragments.
__global__ __launch_bounds__(256)
void k_rgcn(const float* __restrict__ sums, const float* __restrict__ rcnt,
            const float* __restrict__ hin, const unsigned short* __restrict__ wf,
            const float* __restrict__ bias, float* __restrict__ hout) {
  __shared__ unsigned short At[2][16 * LDS_STRIDE];
  const int tid = threadIdx.x, lane = tid & 31, wv = tid >> 5;
  const int n0 = blockIdx.x * 16;
  const int ft0 = wv * 2, ft1 = wv * 2 + 1;

  // staging geometry: each thread stages one K-pair of one row (16x32 tile)
  const int sm  = tid >> 4;            // row 0..15
  const int sk2 = (tid & 15) * 2;      // even kk
  const int ldsu = sm * (LDS_STRIDE / 2) + (tid & 15);   // uint index
  const float* srow = sums + (size_t)(n0 + sm) * RR * DD;
  const float* rrow = rcnt + (n0 + sm) * RR;
  const float* hrow = hin + (size_t)(n0 + sm) * DD;

  auto stage = [&](int kc, int b) {
    float2 v;
    if (kc < NKC_REL) {                         // uniform branch (kc is scalar)
      int r = kc >> 3;
      int d = ((kc & 7) << 5) + sk2;
      v = *(const float2*)&srow[r * DD + d];
      float rc = rrow[r];
      v.x *= rc; v.y *= rc;
    } else {
      v = *(const float2*)&hrow[((kc - NKC_REL) << 5) + sk2];
    }
    ((unsigned*)At[b])[ldsu] = f2bf_pack(v.x, v.y);
  };

  // fragment addressing (A and B: lane half h holds k 16h..16h+15)
  const int a_off = (lane & 15) * LDS_STRIDE + ((lane < 16) ? 0 : 16);
  const uint4* pB0 = ((const uint4*)wf) + (size_t)(ft0 * 32 + lane) * 2;
  const uint4* pB1 = ((const uint4*)wf) + (size_t)(ft1 * 32 + lane) * 2;

  v8f c0 = {}; v8f c1 = {};
  stage(0, 0);
  __syncthreads();
  for (int kc = 0; kc < NKC; ++kc) {
    if (kc + 1 < NKC) stage(kc + 1, (kc + 1) & 1);
    const unsigned short* buf = At[kc & 1];
    FragBF a;
    a.q[0] = *(const uint4*)&buf[a_off];
    a.q[1] = *(const uint4*)&buf[a_off + 8];
    FragBF b0, b1;
    b0.q[0] = pB0[0]; b0.q[1] = pB0[1];
    b1.q[0] = pB1[0]; b1.q[1] = pB1[1];
    pB0 += 1024; pB1 += 1024;                   // next 16KB fragment block
    c0 = __builtin_amdgcn_wmma_f32_16x16x32_bf16(false, a.v, false, b0.v, (short)0, c0, false, false);
    c1 = __builtin_amdgcn_wmma_f32_16x16x32_bf16(false, a.v, false, b1.v, (short)0, c1, false, false);
    __syncthreads();
  }
  // epilogue: bias + relu; C/D layout: col = lane%16, row = v + 8*(lane/16)
  int Nc = lane & 15;
  int f_a = ft0 * 16 + Nc, f_b = ft1 * 16 + Nc;
  float ba = bias[f_a], bb = bias[f_b];
  int mbase = (lane >> 4) * 8;
  #pragma unroll
  for (int v = 0; v < 8; ++v) {
    int row = n0 + mbase + v;
    hout[(size_t)row * DD + f_a] = fmaxf(c0[v] + ba, 0.f);
    hout[(size_t)row * DD + f_b] = fmaxf(c1[v] + bb, 0.f);
  }
}

// ---------------- head: logits = relu(h@w1+b1)@w2 + b2 ----------------
__global__ __launch_bounds__(256)
void k_head(const float* __restrict__ hin, const unsigned short* __restrict__ wf,
            const float* __restrict__ b1, const float* __restrict__ w2,
            const float* __restrict__ b2, float* __restrict__ logits) {
  __shared__ unsigned short At[2][16 * LDS_STRIDE];
  __shared__ float rowacc[16];
  const int tid = threadIdx.x, lane = tid & 31, wv = tid >> 5;
  const int n0 = blockIdx.x * 16;
  const int ft0 = wv * 2, ft1 = wv * 2 + 1;
  if (tid < 16) rowacc[tid] = 0.f;

  const int sm  = tid >> 4;
  const int sk2 = (tid & 15) * 2;
  const int ldsu = sm * (LDS_STRIDE / 2) + (tid & 15);
  const float* hrow = hin + (size_t)(n0 + sm) * DD;
  auto stage = [&](int kc, int b) {
    float2 v = *(const float2*)&hrow[(kc << 5) + sk2];
    ((unsigned*)At[b])[ldsu] = f2bf_pack(v.x, v.y);
  };

  const int a_off = (lane & 15) * LDS_STRIDE + ((lane < 16) ? 0 : 16);
  const uint4* pB0 = ((const uint4*)wf) + (size_t)(ft0 * 32 + lane) * 2;
  const uint4* pB1 = ((const uint4*)wf) + (size_t)(ft1 * 32 + lane) * 2;

  v8f c0 = {}; v8f c1 = {};
  stage(0, 0);
  __syncthreads();
  for (int kc = 0; kc < NKC_HEAD; ++kc) {
    if (kc + 1 < NKC_HEAD) stage(kc + 1, (kc + 1) & 1);
    const unsigned short* buf = At[kc & 1];
    FragBF a;
    a.q[0] = *(const uint4*)&buf[a_off];
    a.q[1] = *(const uint4*)&buf[a_off + 8];
    FragBF b0, b1;
    b0.q[0] = pB0[0]; b0.q[1] = pB0[1];
    b1.q[0] = pB1[0]; b1.q[1] = pB1[1];
    pB0 += 1024; pB1 += 1024;
    c0 = __builtin_amdgcn_wmma_f32_16x16x32_bf16(false, a.v, false, b0.v, (short)0, c0, false, false);
    c1 = __builtin_amdgcn_wmma_f32_16x16x32_bf16(false, a.v, false, b1.v, (short)0, c1, false, false);
    __syncthreads();
  }
  int Nc = lane & 15;
  int f_a = ft0 * 16 + Nc, f_b = ft1 * 16 + Nc;
  float ba = b1[f_a], bb = b1[f_b];
  float wa = w2[f_a], wb = w2[f_b];
  int mbase = (lane >> 4) * 8;
  #pragma unroll
  for (int v = 0; v < 8; ++v) {
    float t = fmaxf(c0[v] + ba, 0.f) * wa + fmaxf(c1[v] + bb, 0.f) * wb;
    atomicAdd(&rowacc[mbase + v], t);               // ds_add_f32 row reduce
  }
  __syncthreads();
  if (tid < 16) logits[n0 + tid] = rowacc[tid] + b2[0];
}

// ---------------- masked softmax ----------------
__global__ void k_mask(const int* __restrict__ ord, float* __restrict__ logits) {
  int i = threadIdx.x;
  if (i < KORD) logits[ord[i]] = -__builtin_inff();
}

__global__ void k_max(const float* __restrict__ x, float* __restrict__ part) {
  __shared__ float s[256];
  float m = -__builtin_inff();
  for (int i = blockIdx.x * 256 + threadIdx.x; i < NN; i += 256 * 256)
    m = fmaxf(m, x[i]);
  s[threadIdx.x] = m; __syncthreads();
  for (int st = 128; st > 0; st >>= 1) {
    if (threadIdx.x < st) s[threadIdx.x] = fmaxf(s[threadIdx.x], s[threadIdx.x + st]);
    __syncthreads();
  }
  if (threadIdx.x == 0) part[blockIdx.x] = s[0];
}

__global__ void k_max2(float* __restrict__ red) {     // red[0..255] -> red[512]
  __shared__ float s[256];
  s[threadIdx.x] = red[threadIdx.x]; __syncthreads();
  for (int st = 128; st > 0; st >>= 1) {
    if (threadIdx.x < st) s[threadIdx.x] = fmaxf(s[threadIdx.x], s[threadIdx.x + st]);
    __syncthreads();
  }
  if (threadIdx.x == 0) red[512] = s[0];
}

__global__ void k_sum(const float* __restrict__ x, float* __restrict__ red) {
  __shared__ float s[256];
  float gm = red[512];
  float a = 0.f;
  for (int i = blockIdx.x * 256 + threadIdx.x; i < NN; i += 256 * 256)
    a += expf(x[i] - gm);
  s[threadIdx.x] = a; __syncthreads();
  for (int st = 128; st > 0; st >>= 1) {
    if (threadIdx.x < st) s[threadIdx.x] += s[threadIdx.x + st];
    __syncthreads();
  }
  if (threadIdx.x == 0) red[256 + blockIdx.x] = s[0];
}

__global__ void k_sum2(float* __restrict__ red) {     // red[256..511] -> red[513]
  __shared__ float s[256];
  s[threadIdx.x] = red[256 + threadIdx.x]; __syncthreads();
  for (int st = 128; st > 0; st >>= 1) {
    if (threadIdx.x < st) s[threadIdx.x] += s[threadIdx.x + st];
    __syncthreads();
  }
  if (threadIdx.x == 0) red[513] = s[0];
}

__global__ void k_norm(const float* __restrict__ x, const float* __restrict__ red,
                       float* __restrict__ out) {
  int i = blockIdx.x * blockDim.x + threadIdx.x;
  if (i >= NN) return;
  out[i] = expf(x[i] - red[512]) / red[513];
}

// ---------------- host orchestration ----------------
extern "C" void kernel_launch(void* const* d_in, const int* in_sizes, int n_in,
                              void* d_out, int out_size, void* d_ws, size_t ws_size,
                              hipStream_t stream) {
  const int*   x      = (const int*)  d_in[0];
  const int*   eidx   = (const int*)  d_in[1];          // [2, E]
  const int*   etype  = (const int*)  d_in[2];
  const int*   ord    = (const int*)  d_in[3];
  const float* emb    = (const float*)d_in[4];
  const float* W_rel  = (const float*)d_in[5];
  const float* W_root = (const float*)d_in[6];
  const float* b_rel  = (const float*)d_in[7];
  const float* w1     = (const float*)d_in[8];
  const float* b1     = (const float*)d_in[9];
  const float* w2     = (const float*)d_in[10];
  const float* b2     = (const float*)d_in[11];
  const float* pos    = (const float*)d_in[12];
  const int* src = eidx;
  const int* dst = eidx + EE;

  char* ws = (char*)d_ws;
  size_t o = 0;
  float* h0 = (float*)(ws + o);               o += (size_t)NN * DD * 4;
  float* h1 = (float*)(ws + o);               o += (size_t)NN * DD * 4;
  float* sums = (float*)(ws + o);             o += (size_t)NN * RR * DD * 4;
  float* cnt = (float*)(ws + o);              o += (size_t)NN * RR * 4;
  unsigned short* wf = (unsigned short*)(ws + o); o += (size_t)TOT_FRAG * 2;
  o = (o + 255) & ~(size_t)255;
  float* logits = (float*)(ws + o);           o += (size_t)NN * 4;
  float* red = (float*)(ws + o);

  // weight repack (bf16 fragments) + initial h
  k_prep<<<TOT_FRAG / 256, 256, 0, stream>>>(W_rel, W_root, w1, wf);
  k_embed<<<(NN * (DD / 4)) / 256, 256, 0, stream>>>(x, (const float4*)emb, (float4*)h0);
  k_posadd<<<(KORD * DD + 255) / 256, 256, 0, stream>>>(ord, pos, h0);

  float* hin = h0;
  float* hout = h1;
  const int sums_n4 = NN * RR * DD / 4;
  const int cnt_n4  = NN * RR / 4;
  for (int l = 0; l < NLAYERS; ++l) {
    k_zero4<<<(sums_n4 + 255) / 256, 256, 0, stream>>>((float4*)sums, sums_n4);
    k_zero4<<<(cnt_n4 + 255) / 256, 256, 0, stream>>>((float4*)cnt, cnt_n4);
    k_count<<<(EE + 255) / 256, 256, 0, stream>>>(dst, etype, cnt);
    k_rcnt<<<(NN * RR + 255) / 256, 256, 0, stream>>>(cnt);
    k_scatter<<<(EE * 32) / 256, 256, 0, stream>>>(src, dst, etype, hin, sums);
    k_rgcn<<<NN / 16, 256, 0, stream>>>(sums, cnt, hin, wf + (size_t)l * PER_LAYER_FRAG,
                                        b_rel + l * DD, hout);
    float* t = hin; hin = hout; hout = t;
  }

  k_head<<<NN / 16, 256, 0, stream>>>(hin, wf + (size_t)NLAYERS * PER_LAYER_FRAG,
                                      b1, w2, b2, logits);
  k_mask<<<1, 64, 0, stream>>>(ord, logits);
  k_max<<<256, 256, 0, stream>>>(logits, red);
  k_max2<<<1, 256, 0, stream>>>(red);
  k_sum<<<256, 256, 0, stream>>>(logits, red);
  k_sum2<<<1, 256, 0, stream>>>(red);
  k_norm<<<(NN + 255) / 256, 256, 0, stream>>>(logits, red, (float*)d_out);
}
